// AttentionBlock_64115271795287
// MI455X (gfx1250) — matrix operand order repaired and checked by reading, not compiled, not run on hardware
//
#include <hip/hip_runtime.h>

#define BATCH 4
#define SQ    2048
#define DIM   768
#define NH    12
#define DH    64
#define FF    1536
#define MTOT  (BATCH * SQ)   // 8192

typedef __attribute__((ext_vector_type(16))) __bf16       v16bf;
typedef __attribute__((ext_vector_type(8)))  float        v8f;
typedef __attribute__((ext_vector_type(4)))  unsigned int v4u;
typedef __attribute__((ext_vector_type(8)))  int          v8i;
typedef __attribute__((ext_vector_type(4)))  int          v4i;

union Frag { v16bf v; uint4 q[2]; };

__device__ __forceinline__ uint4 ldg16(const __bf16* p) {
    return *reinterpret_cast<const uint4*>(p);
}

#if __has_builtin(__builtin_amdgcn_tensor_load_to_lds) && __has_builtin(__builtin_amdgcn_s_wait_tensorcnt)
#define USE_TDM 1
#endif

// ---------------- 16-lane row reductions (DPP butterfly if available) ----------------
#if __has_builtin(__builtin_amdgcn_mov_dpp)
template<int CTRL>
__device__ __forceinline__ float dppf(float x) {
    return __builtin_bit_cast(float,
        __builtin_amdgcn_mov_dpp(__builtin_bit_cast(int, x), CTRL, 0xF, 0xF, true));
}
// Butterfly across a 16-lane DPP row: xor1, xor2 via quad_perm, then group swaps
// via row_half_mirror / row_mirror (valid: values uniform in subgroup after each step).
__device__ __forceinline__ float rmax16(float x) {
    x = fmaxf(x, dppf<0xB1>(x));    // quad_perm [1,0,3,2]
    x = fmaxf(x, dppf<0x4E>(x));    // quad_perm [2,3,0,1]
    x = fmaxf(x, dppf<0x141>(x));   // row_half_mirror
    x = fmaxf(x, dppf<0x140>(x));   // row_mirror
    return x;
}
__device__ __forceinline__ float rsum16(float x) {
    x += dppf<0xB1>(x);
    x += dppf<0x4E>(x);
    x += dppf<0x141>(x);
    x += dppf<0x140>(x);
    return x;
}
#else
__device__ __forceinline__ float rmax16(float x) {
    for (int o = 8; o >= 1; o >>= 1) x = fmaxf(x, __shfl_xor(x, o, 16));
    return x;
}
__device__ __forceinline__ float rsum16(float x) {
    for (int o = 8; o >= 1; o >>= 1) x += __shfl_xor(x, o, 16);
    return x;
}
#endif

// ---------------------------------------------------------------- converts
__global__ __launch_bounds__(256) void f32_to_bf16_kernel(
    const float* __restrict__ src, __bf16* __restrict__ dst, int n)
{
    int i = blockIdx.x * 256 + threadIdx.x;
    if (i < n) dst[i] = (__bf16)src[i];
}

// W is [K][N] row-major f32 -> Wt [N][K] bf16 (GEMM B-fragments contiguous in k)
__global__ __launch_bounds__(256) void transpose_to_bf16_kernel(
    const float* __restrict__ W, __bf16* __restrict__ Wt, int K, int N)
{
    int i = blockIdx.x * 256 + threadIdx.x;
    if (i >= K * N) return;
    int k = i / N, n = i - k * N;
    Wt[(size_t)n * K + k] = (__bf16)W[i];
}

// ---------------------------------------------------------------- WMMA GEMM
// One wave computes a 32x64 tile of C = A[MxK] * Bt^T + bias, then * oscale.
// A: bf16 row-major [M][K]; Bt: bf16 [N][K] (B column-major).
// Weight tile (64 rows x 32 k) is staged into LDS by the Tensor Data Mover
// (double-buffered, s_wait_tensorcnt) when available.
// OUTMODE: 0 = f32 row-major, 1 = bf16 row-major,
//          2 = bf16 V-transposed [BATCH][NH][DH][SQ]  (m->(b,s), n->(h,d))
template<int OUTMODE, bool RELU>
__global__ __launch_bounds__(32) void wmma_gemm_kernel(
    const __bf16* __restrict__ A, const __bf16* __restrict__ Bt,
    const float* __restrict__ bias, void* __restrict__ Cout,
    int M, int N, int K, float oscale)
{
    const int lane = threadIdx.x;
    const int half = lane >> 4;     // 0: lanes 0-15, 1: lanes 16-31
    const int lr   = lane & 15;
    const int m0 = blockIdx.x * 32;
    const int n0 = blockIdx.y * 64;

    const __bf16* arow0 = A + (size_t)(m0 + lr) * K;
    const __bf16* arow1 = A + (size_t)(m0 + 16 + lr) * K;

    v8f acc[2][4];
    #pragma unroll
    for (int mt = 0; mt < 2; ++mt)
        #pragma unroll
        for (int t = 0; t < 4; ++t) acc[mt][t] = v8f{};

#ifdef USE_TDM
    // Double-buffered weight tile: 64 rows x 32 k of bf16 (4 KB per buffer).
    __shared__ __align__(16) __bf16 btile[2][64][32];

    auto tdm_issue = [&](int buf, int k0) {
        const uint64_t ga  = (uint64_t)(uintptr_t)(Bt + (size_t)n0 * K + k0);
        const uint32_t lds = (uint32_t)(uintptr_t)(&btile[buf][0][0]);
        // D# group 0: count=1 | lds_addr | global_addr[56:0] | type=2
        v4u g0 = { 1u, lds, (uint32_t)ga,
                   ((uint32_t)(ga >> 32) & 0x01FFFFFFu) | 0x80000000u };
        // D# group 1: data_size=2B; tensor_dim0=K, tensor_dim1=N;
        // tile_dim0=32 (k), tile_dim1=64 (rows); tensor_dim0_stride=K
        v8i g1 = { (int)0x10000u,
                   (int)((unsigned)K << 16),
                   (int)(((unsigned)N & 0xFFFFu) << 16),
                   (int)(32u << 16),
                   64,
                   (int)K,
                   0, 0 };
        v4i z4 = { 0, 0, 0, 0 };
        v8i z8 = { 0, 0, 0, 0, 0, 0, 0, 0 };
        __builtin_amdgcn_tensor_load_to_lds(g0, g1, z4, z4, z8, 0);
    };

    tdm_issue(0, 0);
#endif

    for (int k0 = 0; k0 < K; k0 += 32) {
        // A fragments (16x32 bf16): lane<16 -> k {0..15}; lane>=16 -> {16..31}
        Frag a0, a1;
        a0.q[0] = ldg16(arow0 + k0 + half * 16);
        a0.q[1] = ldg16(arow0 + k0 + half * 16 + 8);
        a1.q[0] = ldg16(arow1 + k0 + half * 16);
        a1.q[1] = ldg16(arow1 + k0 + half * 16 + 8);
        if (k0 + 64 < K) {
            __builtin_prefetch(arow0 + k0 + 64, 0, 1);
            __builtin_prefetch(arow1 + k0 + 64, 0, 1);
        }

#ifdef USE_TDM
        const int cur = (k0 >> 5) & 1;
        if (k0 + 32 < K) {
            tdm_issue(cur ^ 1, k0 + 32);
            __builtin_amdgcn_s_wait_tensorcnt(1);
        } else {
            __builtin_amdgcn_s_wait_tensorcnt(0);
        }
        #pragma unroll
        for (int t = 0; t < 4; ++t) {
            const __bf16* bp = &btile[cur][t * 16 + lr][half * 16];
            Frag bf;
            bf.q[0] = *reinterpret_cast<const uint4*>(bp);
            bf.q[1] = *reinterpret_cast<const uint4*>(bp + 8);
            acc[0][t] = __builtin_amdgcn_wmma_f32_16x16x32_bf16(
                false, a0.v, false, bf.v, (short)0, acc[0][t], false, false);
            acc[1][t] = __builtin_amdgcn_wmma_f32_16x16x32_bf16(
                false, a1.v, false, bf.v, (short)0, acc[1][t], false, false);
        }
#else
        #pragma unroll
        for (int t = 0; t < 4; ++t) {
            const __bf16* bp = Bt + (size_t)(n0 + t * 16 + lr) * K + k0 + half * 16;
            Frag bf;
            bf.q[0] = ldg16(bp);
            bf.q[1] = ldg16(bp + 8);
            acc[0][t] = __builtin_amdgcn_wmma_f32_16x16x32_bf16(
                false, a0.v, false, bf.v, (short)0, acc[0][t], false, false);
            acc[1][t] = __builtin_amdgcn_wmma_f32_16x16x32_bf16(
                false, a1.v, false, bf.v, (short)0, acc[1][t], false, false);
        }
#endif
    }

    #pragma unroll
    for (int t = 0; t < 4; ++t) {
        const int n = n0 + t * 16 + lr;
        const float bv = bias[n];
        #pragma unroll
        for (int mt = 0; mt < 2; ++mt) {
            #pragma unroll
            for (int r = 0; r < 8; ++r) {
                const int m = m0 + mt * 16 + r + 8 * half;   // C layout: vgpr r, lane half
                float v = (acc[mt][t][r] + bv) * oscale;
                if (RELU) v = fmaxf(v, 0.0f);
                if (OUTMODE == 0) {
                    ((float*)Cout)[(size_t)m * N + n] = v;
                } else if (OUTMODE == 1) {
                    ((__bf16*)Cout)[(size_t)m * N + n] = (__bf16)v;
                } else {
                    const int b = m >> 11, s = m & (SQ - 1);
                    const int h = n >> 6,  d = n & (DH - 1);
                    ((__bf16*)Cout)[(((size_t)b * NH + h) * DH + d) * SQ + s] = (__bf16)v;
                }
            }
        }
    }
}

// ---------------------------------------------------------------- attention
// One wave per (b, h, 16-query tile). Flash-style online softmax, 32 keys/iter.
// 1/sqrt(DH) is pre-folded into Q. Qb, Kb: bf16 [MTOT][DIM]; Vt: bf16 [B][NH][DH][SQ].
__global__ __launch_bounds__(32) void attn_kernel(
    const __bf16* __restrict__ Qb, const __bf16* __restrict__ Kb,
    const __bf16* __restrict__ Vt, __bf16* __restrict__ Ctx)
{
    const int lane = threadIdx.x;
    const int half = lane >> 4;
    const int lr   = lane & 15;

    const int bid = blockIdx.x;
    const int qt  = bid & (SQ / 16 - 1);
    const int h   = (bid >> 7) % NH;
    const int b   = bid / ((SQ / 16) * NH);

    __shared__ __align__(16) __bf16 pt[16][32];

    const __bf16* qbase = Qb + (size_t)(b * SQ + qt * 16 + lr) * DIM + h * DH;
    Frag aq0, aq1;
    aq0.q[0] = ldg16(qbase + 0  + half * 16);
    aq0.q[1] = ldg16(qbase + 8  + half * 16);
    aq1.q[0] = ldg16(qbase + 32 + half * 16);
    aq1.q[1] = ldg16(qbase + 40 + half * 16);

    float Mrow[8], Lrow[8];
    v8f accO[4];
    #pragma unroll
    for (int r = 0; r < 8; ++r) { Mrow[r] = -1e30f; Lrow[r] = 0.0f; }
    accO[0] = v8f{}; accO[1] = v8f{}; accO[2] = v8f{}; accO[3] = v8f{};

    for (int kt = 0; kt < SQ / 32; ++kt) {
        // ---- scores: two 16x16 C tiles (keys kt*32 + {0..15, 16..31})
        v8f s0 = v8f{}, s1 = v8f{};
        {
            const __bf16* kb0 = Kb + (size_t)(b * SQ + kt * 32 + lr) * DIM + h * DH;
            Frag bk;
            bk.q[0] = ldg16(kb0 + half * 16);
            bk.q[1] = ldg16(kb0 + half * 16 + 8);
            s0 = __builtin_amdgcn_wmma_f32_16x16x32_bf16(false, aq0.v, false, bk.v, (short)0, s0, false, false);
            bk.q[0] = ldg16(kb0 + 32 + half * 16);
            bk.q[1] = ldg16(kb0 + 32 + half * 16 + 8);
            s0 = __builtin_amdgcn_wmma_f32_16x16x32_bf16(false, aq1.v, false, bk.v, (short)0, s0, false, false);

            const __bf16* kb1 = kb0 + (size_t)16 * DIM;
            bk.q[0] = ldg16(kb1 + half * 16);
            bk.q[1] = ldg16(kb1 + half * 16 + 8);
            s1 = __builtin_amdgcn_wmma_f32_16x16x32_bf16(false, aq0.v, false, bk.v, (short)0, s1, false, false);
            bk.q[0] = ldg16(kb1 + 32 + half * 16);
            bk.q[1] = ldg16(kb1 + 32 + half * 16 + 8);
            s1 = __builtin_amdgcn_wmma_f32_16x16x32_bf16(false, aq1.v, false, bk.v, (short)0, s1, false, false);
        }

        // ---- online softmax per row (row m = r + 8*half lives in one 16-lane group)
        float p0[8], p1[8], rsc[8];
        #pragma unroll
        for (int r = 0; r < 8; ++r) {
            const float a0 = s0[r];
            const float a1 = s1[r];
            const float mx = rmax16(fmaxf(a0, a1));
            const float mnew = fmaxf(Mrow[r], mx);
            rsc[r] = __expf(Mrow[r] - mnew);
            const float e0 = __expf(a0 - mnew);
            const float e1 = __expf(a1 - mnew);
            const float sum = rsum16(e0 + e1);
            Lrow[r] = Lrow[r] * rsc[r] + sum;
            Mrow[r] = mnew;
            p0[r] = e0; p1[r] = e1;
        }

        // ---- reshape P: C-layout -> A-layout via LDS
        __syncthreads();
        #pragma unroll
        for (int r = 0; r < 8; ++r) {
            const int m = r + 8 * half;
            pt[m][lr]      = (__bf16)p0[r];
            pt[m][16 + lr] = (__bf16)p1[r];
        }
        __syncthreads();
        Frag pa;
        pa.q[0] = *reinterpret_cast<const uint4*>(&pt[lr][half * 16]);
        pa.q[1] = *reinterpret_cast<const uint4*>(&pt[lr][half * 16 + 8]);

        // ---- rescale O, then accumulate P (16x32) @ V-tile (32x64)
        #pragma unroll
        for (int t = 0; t < 4; ++t) {
            #pragma unroll
            for (int r = 0; r < 8; ++r) accO[t][r] *= rsc[r];
        }
        #pragma unroll
        for (int t = 0; t < 4; ++t) {
            const __bf16* vb = Vt + (((size_t)b * NH + h) * DH + t * 16 + lr) * SQ
                                  + kt * 32 + half * 16;
            Frag bv;
            bv.q[0] = ldg16(vb);
            bv.q[1] = ldg16(vb + 8);
            accO[t] = __builtin_amdgcn_wmma_f32_16x16x32_bf16(
                false, pa.v, false, bv.v, (short)0, accO[t], false, false);
        }
    }

    // ---- normalize and write ctx (bf16, [MTOT][DIM] layout for next GEMM)
    float rinv[8];
    #pragma unroll
    for (int r = 0; r < 8; ++r) rinv[r] = 1.0f / Lrow[r];
    #pragma unroll
    for (int t = 0; t < 4; ++t) {
        #pragma unroll
        for (int r = 0; r < 8; ++r) {
            const int m = r + 8 * half;
            const int row = b * SQ + qt * 16 + m;
            const int col = h * DH + t * 16 + lr;
            Ctx[(size_t)row * DIM + col] = (__bf16)(accO[t][r] * rinv[r]);
        }
    }
}

// ---------------------------------------------------------------- residual + LayerNorm
__global__ __launch_bounds__(256) void add_ln_kernel(
    const float* __restrict__ x, const float* __restrict__ o2,
    const float* __restrict__ gamma, const float* __restrict__ beta,
    float* __restrict__ out)
{
    const int row = blockIdx.x;
    const int tid = threadIdx.x;
    __shared__ float red[256];

    const float* xr  = x  + (size_t)row * DIM;
    const float* orw = o2 + (size_t)row * DIM;

    float v[3];
    float s = 0.0f;
    #pragma unroll
    for (int i = 0; i < 3; ++i) {
        const int c = tid + i * 256;
        v[i] = xr[c] + orw[c];
        s += v[i];
    }
    red[tid] = s; __syncthreads();
    for (int off = 128; off > 0; off >>= 1) {
        if (tid < off) red[tid] += red[tid + off];
        __syncthreads();
    }
    const float mu = red[0] * (1.0f / DIM);
    __syncthreads();

    float s2 = 0.0f;
    #pragma unroll
    for (int i = 0; i < 3; ++i) { const float d = v[i] - mu; s2 += d * d; }
    red[tid] = s2; __syncthreads();
    for (int off = 128; off > 0; off >>= 1) {
        if (tid < off) red[tid] += red[tid + off];
        __syncthreads();
    }
    const float inv = rsqrtf(red[0] * (1.0f / DIM) + 1e-5f);

    #pragma unroll
    for (int i = 0; i < 3; ++i) {
        const int c = tid + i * 256;
        out[(size_t)row * DIM + c] = (v[i] - mu) * inv * gamma[c] + beta[c];
    }
}

// ---------------------------------------------------------------- launch
extern "C" void kernel_launch(void* const* d_in, const int* in_sizes, int n_in,
                              void* d_out, int out_size, void* d_ws, size_t ws_size,
                              hipStream_t stream) {
    const float* x     = (const float*)d_in[0];
    const float* Wq    = (const float*)d_in[1];
    const float* bq    = (const float*)d_in[2];
    const float* Wk    = (const float*)d_in[3];
    const float* bk    = (const float*)d_in[4];
    const float* Wv    = (const float*)d_in[5];
    const float* bv    = (const float*)d_in[6];
    const float* W1    = (const float*)d_in[7];
    const float* b1    = (const float*)d_in[8];
    const float* W2    = (const float*)d_in[9];
    const float* b2    = (const float*)d_in[10];
    const float* gamma = (const float*)d_in[11];
    const float* beta  = (const float*)d_in[12];
    float* out = (float*)d_out;

    char* ws = (char*)d_ws;
    size_t off = 0;
    auto carve = [&](size_t bytes) -> char* {
        char* p = ws + off;
        off += (bytes + 255) & ~(size_t)255;
        return p;
    };
    __bf16* Xb  = (__bf16*)carve((size_t)MTOT * DIM * 2);
    __bf16* Wqt = (__bf16*)carve((size_t)DIM * DIM * 2);
    __bf16* Wkt = (__bf16*)carve((size_t)DIM * DIM * 2);
    __bf16* Wvt = (__bf16*)carve((size_t)DIM * DIM * 2);
    __bf16* W1t = (__bf16*)carve((size_t)FF * DIM * 2);
    __bf16* W2t = (__bf16*)carve((size_t)DIM * FF * 2);
    __bf16* Qb  = (__bf16*)carve((size_t)MTOT * DIM * 2);
    __bf16* Kb  = (__bf16*)carve((size_t)MTOT * DIM * 2);
    __bf16* Vtb = (__bf16*)carve((size_t)BATCH * NH * DH * SQ * 2);
    __bf16* Ctx = (__bf16*)carve((size_t)MTOT * DIM * 2);
    __bf16* H1b = (__bf16*)carve((size_t)MTOT * FF * 2);
    float*  O2f = (float*) carve((size_t)MTOT * DIM * 4);
    (void)ws_size; (void)n_in; (void)in_sizes; (void)out_size;

    // 1) converts
    {
        int n = MTOT * DIM;
        f32_to_bf16_kernel<<<(n + 255) / 256, 256, 0, stream>>>(x, Xb, n);
        int nw = DIM * DIM;
        transpose_to_bf16_kernel<<<(nw + 255) / 256, 256, 0, stream>>>(Wq, Wqt, DIM, DIM);
        transpose_to_bf16_kernel<<<(nw + 255) / 256, 256, 0, stream>>>(Wk, Wkt, DIM, DIM);
        transpose_to_bf16_kernel<<<(nw + 255) / 256, 256, 0, stream>>>(Wv, Wvt, DIM, DIM);
        int n1 = DIM * FF;
        transpose_to_bf16_kernel<<<(n1 + 255) / 256, 256, 0, stream>>>(W1, W1t, DIM, FF);
        transpose_to_bf16_kernel<<<(n1 + 255) / 256, 256, 0, stream>>>(W2, W2t, FF, DIM);
    }

    // 2) QKV projections (bf16 WMMA, f32 accumulate); softmax scale folded into Q
    {
        dim3 g(MTOT / 32, DIM / 64);
        wmma_gemm_kernel<1, false><<<g, 32, 0, stream>>>(Xb, Wqt, bq, Qb,  MTOT, DIM, DIM, 0.125f);
        wmma_gemm_kernel<1, false><<<g, 32, 0, stream>>>(Xb, Wkt, bk, Kb,  MTOT, DIM, DIM, 1.0f);
        wmma_gemm_kernel<2, false><<<g, 32, 0, stream>>>(Xb, Wvt, bv, Vtb, MTOT, DIM, DIM, 1.0f);
    }

    // 3) fused flash attention
    attn_kernel<<<BATCH * NH * (SQ / 16), 32, 0, stream>>>(Qb, Kb, Vtb, Ctx);

    // 4) MLP
    {
        dim3 g1(MTOT / 32, FF / 64);
        wmma_gemm_kernel<1, true><<<g1, 32, 0, stream>>>(Ctx, W1t, b1, H1b, MTOT, FF, DIM, 1.0f);
        dim3 g2(MTOT / 32, DIM / 64);
        wmma_gemm_kernel<0, false><<<g2, 32, 0, stream>>>(H1b, W2t, b2, O2f, MTOT, DIM, FF, 1.0f);
    }

    // 5) residual + LayerNorm (fp32)
    add_ln_kernel<<<MTOT, 256, 0, stream>>>(x, O2f, gamma, beta, out);
}
